// GCNModel_54795192762716
// MI455X (gfx1250) — matrix operand order repaired and checked by reading, not compiled, not run on hardware
//
#include <hip/hip_runtime.h>

// ---------------------------------------------------------------------------
// GCN (3x GCNConv + 3x MLP) for MI455X / gfx1250, wave32, WMMA bf16.
// GEMM: single-buffered LDS B panel (ds_load feeds WMMA), fully unrolled K.
// Aggregation: CSR gather (no f32 atomics).
// ---------------------------------------------------------------------------

typedef __attribute__((ext_vector_type(16))) __bf16 v16bf;
typedef __attribute__((ext_vector_type(8)))  __bf16 v8bf;
typedef __attribute__((ext_vector_type(8)))  float  v8f;

#define NNODES 20000
#define NEDGES 320000
#define DIN    300
#define KPAD0  320
#define EMB    256

// --------------------------- elementwise / packing -------------------------

// fp32 [N, Din] -> bf16 [N, Kpad], zero padded in K
__global__ void __launch_bounds__(256)
cvt_pad_kernel(const float* __restrict__ x, __bf16* __restrict__ xbf,
               int n, int din, int kpad) {
  int idx = blockIdx.x * blockDim.x + threadIdx.x;
  if (idx >= n * kpad) return;
  int r = idx / kpad, c = idx - r * kpad;
  float v = (c < din) ? x[(size_t)r * din + c] : 0.0f;
  xbf[idx] = (__bf16)v;
}

// Pack W [K x 256] (row-major fp32) into wave32 WMMA B-operand tiles (bf16).
// Tile t = kt*16 + nt; per tile 32 lanes x 16 bf16 contiguous.
// Lane l, element j:  k = kt*32 + (l>>4)*16 + j,  n = nt*16 + (l&15).
// For fixed kt the 16 tiles (16KB) are contiguous -> one LDS panel slice.
__global__ void __launch_bounds__(256)
pack_w_kernel(const float* __restrict__ W, __bf16* __restrict__ out,
              int K, int Kpad) {
  int idx = blockIdx.x * blockDim.x + threadIdx.x;
  int total = Kpad * EMB;
  if (idx >= total) return;
  int j    = idx & 15;
  int lane = (idx >> 4) & 31;
  int t    = idx >> 9;
  int nt   = t & 15;
  int kt   = t >> 4;
  int k = kt * 32 + (lane >> 4) * 16 + j;
  int n = nt * 16 + (lane & 15);
  float v = (k < K) ? W[(size_t)k * EMB + n] : 0.0f;
  out[idx] = (__bf16)v;
}

// --------------------------- CSR construction ------------------------------

__global__ void __launch_bounds__(256)
zero_int_kernel(int* __restrict__ p, int n) {
  int i = blockIdx.x * blockDim.x + threadIdx.x;
  if (i < n) p[i] = 0;
}

__global__ void __launch_bounds__(256)
hist_kernel(const long long* __restrict__ dst, int* __restrict__ deg, int e) {
  int i = blockIdx.x * blockDim.x + threadIdx.x;
  if (i < e) atomicAdd(&deg[(int)dst[i]], 1);
}

// dinv[i] = rsqrt(in_degree + 1 self loop)
__global__ void __launch_bounds__(256)
dinv_kernel(const int* __restrict__ deg, float* __restrict__ dinv, int n) {
  int i = blockIdx.x * blockDim.x + threadIdx.x;
  if (i >= n) return;
  dinv[i] = rsqrtf((float)(deg[i] + 1));
}

// Per-256-block exclusive scan (Hillis-Steele in LDS); emits block sums.
__global__ void __launch_bounds__(256)
scan_block_kernel(const int* __restrict__ in, int* __restrict__ out,
                  int* __restrict__ sums, int n) {
  __shared__ int tmp[256];
  int tid = threadIdx.x;
  int gid = blockIdx.x * 256 + tid;
  int v = (gid < n) ? in[gid] : 0;
  tmp[tid] = v;
  __syncthreads();
  for (int off = 1; off < 256; off <<= 1) {
    int t = (tid >= off) ? tmp[tid - off] : 0;
    __syncthreads();
    tmp[tid] += t;
    __syncthreads();
  }
  if (gid < n) out[gid] = tmp[tid] - v;          // exclusive
  if (tid == 255) sums[blockIdx.x] = tmp[255];   // inclusive block total
}

// Exclusive scan of up to 128 block sums, in place (single block).
__global__ void __launch_bounds__(128)
scan_sums_kernel(int* __restrict__ sums, int nb) {
  __shared__ int tmp[128];
  int tid = threadIdx.x;
  int v = (tid < nb) ? sums[tid] : 0;
  tmp[tid] = v;
  __syncthreads();
  for (int off = 1; off < 128; off <<= 1) {
    int t = (tid >= off) ? tmp[tid - off] : 0;
    __syncthreads();
    tmp[tid] += t;
    __syncthreads();
  }
  if (tid < nb) sums[tid] = tmp[tid] - v;        // exclusive
}

__global__ void __launch_bounds__(256)
add_offsets_kernel(int* __restrict__ out, const int* __restrict__ sums, int n) {
  int gid = blockIdx.x * 256 + threadIdx.x;
  if (gid < n) out[gid] += sums[blockIdx.x];
}

// cursor[i] = row[i]; row[n] = E  (launch n+1 threads)
__global__ void __launch_bounds__(256)
prep_fill_kernel(int* __restrict__ row, int* __restrict__ cursor, int n, int e) {
  int gid = blockIdx.x * 256 + threadIdx.x;
  if (gid < n) cursor[gid] = row[gid];
  else if (gid == n) row[n] = e;
}

__global__ void __launch_bounds__(256)
fill_kernel(const long long* __restrict__ src, const long long* __restrict__ dst,
            int* __restrict__ cursor, int* __restrict__ col, int e) {
  int i = blockIdx.x * blockDim.x + threadIdx.x;
  if (i >= e) return;
  int d = (int)dst[i];
  int p = atomicAdd(&cursor[d], 1);
  col[p] = (int)src[i];
}

// --------------------------- CSR gather aggregation ------------------------
// One wave per node; lane covers 8 columns.
// obf[i,c] = relu?( bias[c] + dinv[i] * (dinv[i]*h[i,c] + sum_e dinv[src]*h[src,c]) )
__global__ void __launch_bounds__(256)
gather_kernel(const float* __restrict__ h, const float* __restrict__ dinv,
              const float* __restrict__ bias,
              const int* __restrict__ row, const int* __restrict__ col,
              __bf16* __restrict__ obf, int n, int relu) {
  int w = threadIdx.x >> 5;
  int lane = threadIdx.x & 31;
  int i = blockIdx.x * 8 + w;
  if (i >= n) return;
  int c0 = lane * 8;
  float di = dinv[i];

  const float* hi = h + (size_t)i * EMB + c0;
  float4 s0 = *(const float4*)(hi);
  float4 s1 = *(const float4*)(hi + 4);
  s0.x *= di; s0.y *= di; s0.z *= di; s0.w *= di;
  s1.x *= di; s1.y *= di; s1.z *= di; s1.w *= di;

  int beg = row[i], end = row[i + 1];
  for (int j = beg; j < end; ++j) {
    int sidx = col[j];
    float ds = dinv[sidx];
    const float* hs = h + (size_t)sidx * EMB + c0;
    float4 v0 = *(const float4*)(hs);
    float4 v1 = *(const float4*)(hs + 4);
    s0.x = fmaf(ds, v0.x, s0.x); s0.y = fmaf(ds, v0.y, s0.y);
    s0.z = fmaf(ds, v0.z, s0.z); s0.w = fmaf(ds, v0.w, s0.w);
    s1.x = fmaf(ds, v1.x, s1.x); s1.y = fmaf(ds, v1.y, s1.y);
    s1.z = fmaf(ds, v1.z, s1.z); s1.w = fmaf(ds, v1.w, s1.w);
  }

  float4 b0 = *(const float4*)(bias + c0);
  float4 b1 = *(const float4*)(bias + c0 + 4);
  float o[8];
  o[0] = b0.x + di * s0.x; o[1] = b0.y + di * s0.y;
  o[2] = b0.z + di * s0.z; o[3] = b0.w + di * s0.w;
  o[4] = b1.x + di * s1.x; o[5] = b1.y + di * s1.y;
  o[6] = b1.z + di * s1.z; o[7] = b1.w + di * s1.w;

  v8bf pk;
  for (int t = 0; t < 8; ++t) {
    float v = o[t];
    if (relu) v = fmaxf(v, 0.0f);
    pk[t] = (__bf16)v;
  }
  *(v8bf*)(obf + (size_t)i * EMB + c0) = pk;
}

// --------------------------- WMMA GEMM -------------------------------------
// D[M,256] = A[M,lda](bf16 row-major) x Bpk (packed bf16) (+bias)(+relu)
// Block = 8 waves = 32 rows x 256 cols. Per K-step (32), the 16KB B slice is
// staged in a single LDS buffer (short-lived copy regs -> no spills), then
// the 4 WMMAs read B via ds_load_b128. K-loop fully unrolled (KT template).
// REQUIREMENT: grid covers M exactly (M % 32 == 0) -> EXEC all-1s at WMMAs.
template <int KT>
__global__ void __launch_bounds__(256)
gemm_bf16_wmma(const __bf16* __restrict__ A, int lda,
               const __bf16* __restrict__ Bpk,
               float* __restrict__ C,
               const float* __restrict__ bias,
               __bf16* __restrict__ Obf,
               int relu) {
  __shared__ __align__(16) __bf16 smem[8192];     // 16KB B slice

  int tid  = threadIdx.x;
  int lane = tid & 31;
  int w    = tid >> 5;
  int mt   = blockIdx.x * 2 + (w >> 2);   // 16-row tile
  int ng   = w & 3;                       // 64-col group

  int half = lane >> 4;
  int mrow = lane & 15;
  const __bf16* Ap = A + (size_t)(mt * 16 + mrow) * lda + half * 16;
  const uint4* bsrc = (const uint4*)Bpk;  // 16B units; 1024 per K-step slice
  const __bf16* bb  = &smem[(size_t)(ng * 4) * 512 + lane * 16];

  v8f acc0 = {}, acc1 = {}, acc2 = {}, acc3 = {};
  union AV { v16bf v; v8bf h[2]; };

#pragma unroll
  for (int kt = 0; kt < KT; ++kt) {
    // A operand for this K-step (global; overlaps the copy/barrier below)
    AV a;
    a.h[0] = *(const v8bf*)(Ap + kt * 32);
    a.h[1] = *(const v8bf*)(Ap + kt * 32 + 8);

    // cooperative copy of B slice kt -> LDS (short-lived registers)
    {
      const uint4* s = bsrc + (size_t)kt * 1024;
      uint4 t0 = s[tid];
      uint4 t1 = s[256 + tid];
      uint4 t2 = s[512 + tid];
      uint4 t3 = s[768 + tid];
      ((uint4*)smem)[tid]       = t0;
      ((uint4*)smem)[256 + tid] = t1;
      ((uint4*)smem)[512 + tid] = t2;
      ((uint4*)smem)[768 + tid] = t3;
    }
    __syncthreads();

    v16bf b0 = *(const v16bf*)(bb);
    v16bf b1 = *(const v16bf*)(bb + 1 * 512);
    v16bf b2 = *(const v16bf*)(bb + 2 * 512);
    v16bf b3 = *(const v16bf*)(bb + 3 * 512);
    acc0 = __builtin_amdgcn_wmma_f32_16x16x32_bf16(false, a.v, false, b0, (short)0, acc0, false, false);
    acc1 = __builtin_amdgcn_wmma_f32_16x16x32_bf16(false, a.v, false, b1, (short)0, acc1, false, false);
    acc2 = __builtin_amdgcn_wmma_f32_16x16x32_bf16(false, a.v, false, b2, (short)0, acc2, false, false);
    acc3 = __builtin_amdgcn_wmma_f32_16x16x32_bf16(false, a.v, false, b3, (short)0, acc3, false, false);
    __syncthreads();   // all reads done before next K-step overwrites smem
  }

  // C/D layout: lane&15 = N within tile; VGPR r -> M = half*8 + r
  int row0 = mt * 16 + half * 8;
  v8f accs[4] = {acc0, acc1, acc2, acc3};
  for (int j = 0; j < 4; ++j) {
    int colc = ng * 64 + j * 16 + (lane & 15);
    float bv = bias ? bias[colc] : 0.0f;
    for (int r = 0; r < 8; ++r) {
      float v = accs[j][r] + bv;
      if (relu) v = fmaxf(v, 0.0f);
      size_t o = (size_t)(row0 + r) * EMB + colc;
      if (C)   C[o] = v;
      if (Obf) Obf[o] = (__bf16)v;
    }
  }
}

// --------------------------- host launcher ---------------------------------

extern "C" void kernel_launch(void* const* d_in, const int* in_sizes, int n_in,
                              void* d_out, int out_size, void* d_ws, size_t ws_size,
                              hipStream_t stream) {
  const float* x        = (const float*)d_in[0];
  const long long* ei   = (const long long*)d_in[1];   // int64 [2, E]
  const long long* srcp = ei;
  const long long* dstp = ei + NEDGES;
  const float* conv0_w = (const float*)d_in[2];  const float* conv0_b = (const float*)d_in[3];
  const float* conv1_w = (const float*)d_in[4];  const float* conv1_b = (const float*)d_in[5];
  const float* conv2_w = (const float*)d_in[6];  const float* conv2_b = (const float*)d_in[7];
  const float* mlp1_w  = (const float*)d_in[8];  const float* mlp1_b  = (const float*)d_in[9];
  const float* mlp2_w  = (const float*)d_in[10]; const float* mlp2_b  = (const float*)d_in[11];
  const float* mlp3_w  = (const float*)d_in[12]; const float* mlp3_b  = (const float*)d_in[13];
  float* out = (float*)d_out;

  char* ws = (char*)d_ws;
  size_t off = 0;
  auto carve = [&](size_t bytes) -> void* {
    void* p = ws + off;
    off += (bytes + 255) & ~(size_t)255;
    return p;
  };
  __bf16* xbf   = (__bf16*)carve((size_t)NNODES * KPAD0 * 2);
  __bf16* hbfA  = (__bf16*)carve((size_t)NNODES * EMB * 2);
  __bf16* hbfB  = (__bf16*)carve((size_t)NNODES * EMB * 2);
  float*  g0    = (float*) carve((size_t)NNODES * EMB * 4);
  float*  dinv  = (float*) carve((size_t)NNODES * 4);
  int*    deg32 = (int*)   carve((size_t)NNODES * 4);
  int*    row   = (int*)   carve((size_t)(NNODES + 1) * 4);
  int*    cur   = (int*)   carve((size_t)NNODES * 4);
  int*    col   = (int*)   carve((size_t)NEDGES * 4);
  int*    bsums = (int*)   carve((size_t)128 * 4);
  __bf16* w0p   = (__bf16*)carve((size_t)KPAD0 * EMB * 2);
  __bf16* w1p   = (__bf16*)carve((size_t)EMB * EMB * 2);
  __bf16* w2p   = (__bf16*)carve((size_t)EMB * EMB * 2);
  __bf16* w3p   = (__bf16*)carve((size_t)EMB * EMB * 2);
  __bf16* w4p   = (__bf16*)carve((size_t)EMB * EMB * 2);
  __bf16* w5p   = (__bf16*)carve((size_t)EMB * EMB * 2);

  const int B = 256;
  auto blocks = [](long long total, int b) { return (int)((total + b - 1) / b); };
  const int scan_blocks = blocks(NNODES, 256);   // 79

  // ---- feature / weight prep ----
  cvt_pad_kernel<<<blocks((long long)NNODES * KPAD0, B), B, 0, stream>>>(x, xbf, NNODES, DIN, KPAD0);
  pack_w_kernel<<<blocks((long long)KPAD0 * EMB, B), B, 0, stream>>>(conv0_w, w0p, DIN, KPAD0);
  pack_w_kernel<<<blocks((long long)EMB * EMB, B), B, 0, stream>>>(conv1_w, w1p, EMB, EMB);
  pack_w_kernel<<<blocks((long long)EMB * EMB, B), B, 0, stream>>>(conv2_w, w2p, EMB, EMB);
  pack_w_kernel<<<blocks((long long)EMB * EMB, B), B, 0, stream>>>(mlp1_w,  w3p, EMB, EMB);
  pack_w_kernel<<<blocks((long long)EMB * EMB, B), B, 0, stream>>>(mlp2_w,  w4p, EMB, EMB);
  pack_w_kernel<<<blocks((long long)EMB * EMB, B), B, 0, stream>>>(mlp3_w,  w5p, EMB, EMB);

  // ---- CSR build (once; reused by all 3 conv layers) ----
  zero_int_kernel<<<blocks(NNODES, B), B, 0, stream>>>(deg32, NNODES);
  hist_kernel<<<blocks(NEDGES, B), B, 0, stream>>>(dstp, deg32, NEDGES);
  dinv_kernel<<<blocks(NNODES, B), B, 0, stream>>>(deg32, dinv, NNODES);
  scan_block_kernel<<<scan_blocks, 256, 0, stream>>>(deg32, row, bsums, NNODES);
  scan_sums_kernel<<<1, 128, 0, stream>>>(bsums, scan_blocks);
  add_offsets_kernel<<<scan_blocks, 256, 0, stream>>>(row, bsums, NNODES);
  prep_fill_kernel<<<blocks(NNODES + 1, B), B, 0, stream>>>(row, cur, NNODES, NEDGES);
  fill_kernel<<<blocks(NEDGES, B), B, 0, stream>>>(srcp, dstp, cur, col, NEDGES);

  const int gemm_blocks = NNODES / 32;             // 625 (32 rows per block)
  const int gather_g    = blocks(NNODES, 8);       // one wave per node

  // ---- conv0 ----
  gemm_bf16_wmma<KPAD0 / 32><<<gemm_blocks, B, 0, stream>>>(xbf, KPAD0, w0p, g0, nullptr, nullptr, 0);
  gather_kernel<<<gather_g, B, 0, stream>>>(g0, dinv, conv0_b, row, col, hbfA, NNODES, 1);
  // ---- conv1 ----
  gemm_bf16_wmma<EMB / 32><<<gemm_blocks, B, 0, stream>>>(hbfA, EMB, w1p, g0, nullptr, nullptr, 0);
  gather_kernel<<<gather_g, B, 0, stream>>>(g0, dinv, conv1_b, row, col, hbfB, NNODES, 1);
  // ---- conv2 (no relu) ----
  gemm_bf16_wmma<EMB / 32><<<gemm_blocks, B, 0, stream>>>(hbfB, EMB, w2p, g0, nullptr, nullptr, 0);
  gather_kernel<<<gather_g, B, 0, stream>>>(g0, dinv, conv2_b, row, col, hbfA, NNODES, 0);

  // ---- MLP (bias+relu fused in GEMM epilogue; bf16-only intermediates) ----
  gemm_bf16_wmma<EMB / 32><<<gemm_blocks, B, 0, stream>>>(hbfA, EMB, w3p, nullptr, mlp1_b, hbfB, 1);
  gemm_bf16_wmma<EMB / 32><<<gemm_blocks, B, 0, stream>>>(hbfB, EMB, w4p, nullptr, mlp2_b, hbfA, 1);
  gemm_bf16_wmma<EMB / 32><<<gemm_blocks, B, 0, stream>>>(hbfA, EMB, w5p, out, mlp3_b, nullptr, 0);

  (void)in_sizes; (void)n_in; (void)out_size; (void)ws_size;
}
